// VectorDiscretizer_52097953300713
// MI455X (gfx1250) — matrix operand order repaired and checked by reading, not compiled, not run on hardware
//
#include <hip/hip_runtime.h>

#define M_TOK   16384   // B*L
#define N_EMB   8192
#define DIM     1024
#define RPB     128     // rows per block (4 waves x 32 rows)
#define BTHREADS 128
#define CT      16      // codes per tile
#define NTILES  (N_EMB / CT)   // 512
#define A_BYTES (RPB * DIM * 2)        // 262144
#define B_BYTES (CT * DIM * 2)         // 32768
// A panel + double-buffered B tile = 320 KB = full CDNA5 WGP LDS allocation
#define LDS_BYTES (A_BYTES + 2 * B_BYTES)

typedef __attribute__((ext_vector_type(16))) __bf16 v16bf;
typedef __attribute__((ext_vector_type(8)))  __bf16 v8bf;
typedef __attribute__((ext_vector_type(8)))  float  v8f;

static __device__ __forceinline__ unsigned short f32_to_bf16(float f) {
  unsigned u = __float_as_uint(f);
  u += 0x7FFFu + ((u >> 16) & 1u);          // round-to-nearest-even
  return (unsigned short)(u >> 16);
}

// ---------------------------------------------------------------- prep kernels
__global__ __launch_bounds__(256) void conv_z_kernel(
    const float* __restrict__ z, unsigned short* __restrict__ zb, int n) {
  int i = blockIdx.x * 256 + threadIdx.x;
  int stride = gridDim.x * 256;
  for (; i < n; i += stride) zb[i] = f32_to_bf16(z[i]);
}

__global__ __launch_bounds__(256) void conv_e_kernel(
    const float* __restrict__ e, unsigned short* __restrict__ eb,
    float* __restrict__ enorm) {
  const int nrow = blockIdx.x;
  const float* er = e + (size_t)nrow * DIM;
  float s = 0.f;
  for (int j = threadIdx.x; j < DIM; j += 256) {
    float v = er[j];
    eb[(size_t)nrow * DIM + j] = f32_to_bf16(v);
    s += v * v;
  }
  #pragma unroll
  for (int m = 16; m >= 1; m >>= 1) s += __shfl_xor(s, m, 32);
  __shared__ float red[8];
  if ((threadIdx.x & 31) == 0) red[threadIdx.x >> 5] = s;
  __syncthreads();
  if (threadIdx.x == 0) {
    float t = 0.f;
    #pragma unroll
    for (int i = 0; i < 8; ++i) t += red[i];
    enorm[nrow] = t;
  }
}

// ------------------------------------------------- argmin GEMM (WMMA + async)
__global__ __launch_bounds__(BTHREADS) void vq_argmin_kernel(
    const unsigned short* __restrict__ zb,   // [M, D] bf16 bits
    const unsigned short* __restrict__ eb,   // [N, D] bf16 bits
    const float* __restrict__ enorm,         // [N]
    int* __restrict__ idx)                   // [M]
{
  extern __shared__ unsigned short smem[];   // 320 KB: A panel + 2 B tiles
  const int tid  = threadIdx.x;
  const int lane = tid & 31;
  const int wave = tid >> 5;                 // 0..3, owns 32 rows
  const int rowBase = blockIdx.x * RPB;

  const unsigned long long zbase = (unsigned long long)zb;
  const unsigned long long ebase = (unsigned long long)eb;

  const unsigned sB_off0 = (unsigned)A_BYTES;
  const unsigned sB_off1 = (unsigned)A_BYTES + (unsigned)B_BYTES;

  // ---- stage A panel (128 rows x 1024 halves = 256 KB) via async LDS loads --
  {
    const unsigned gA = (unsigned)rowBase * (DIM * 2u);
    #pragma unroll 4
    for (int i = 0; i < 128; ++i) {          // 128 * 128thr * 16B = 256 KB
      unsigned lin  = (unsigned)(i * (BTHREADS * 16) + tid * 16);
      unsigned gofs = gA + lin;
      asm volatile("global_load_async_to_lds_b128 %0, %1, %2"
                   :: "v"(lin), "v"(gofs), "s"(zbase) : "memory");
    }
  }
  // ---- stage B tile 0 (16 codes x 1024 halves = 32 KB) ----
  #pragma unroll 4
  for (int i = 0; i < 16; ++i) {             // 16 * 128thr * 16B = 32 KB
    unsigned lin = (unsigned)(i * (BTHREADS * 16) + tid * 16);
    asm volatile("global_load_async_to_lds_b128 %0, %1, %2"
                 :: "v"(sB_off0 + lin), "v"(lin), "s"(ebase) : "memory");
  }

  float best0[8], best1[8];
  int   bidx0[8], bidx1[8];
  #pragma unroll
  for (int v = 0; v < 8; ++v) {
    best0[v] = 3.4e38f; bidx0[v] = 0;
    best1[v] = 3.4e38f; bidx1[v] = 0;
  }

  const int r  = lane & 15;           // A row-in-tile / B column / result N
  const int kh = (lane >> 4) * 8;     // A lane half-select (ISA 16-bit A layout)
  const int kc = (lane >> 4) * 8;     // B lane half-select (same K order as A)
  const unsigned aRow0Half = (unsigned)((wave * 32 + r) * DIM);        // rows +0..15
  const unsigned aRow1Half = (unsigned)((wave * 32 + 16 + r) * DIM);   // rows +16..31
  const unsigned bColHalf  = (unsigned)(r * DIM);

  for (int ct = 0; ct < NTILES; ++ct) {
    const unsigned curB = (ct & 1) ? sB_off1 : sB_off0;
    if (ct + 1 < NTILES) {
      const unsigned nxtB  = (ct & 1) ? sB_off0 : sB_off1;
      const unsigned gbase = (unsigned)(ct + 1) * (unsigned)B_BYTES;
      #pragma unroll 4
      for (int i = 0; i < 16; ++i) {
        unsigned lin = (unsigned)(i * (BTHREADS * 16) + tid * 16);
        asm volatile("global_load_async_to_lds_b128 %0, %1, %2"
                     :: "v"(nxtB + lin), "v"(gbase + lin), "s"(ebase) : "memory");
      }
      // 16 newer per-wave async ops outstanding -> everything older (tile ct) done
      asm volatile("s_wait_asynccnt 0x10" ::: "memory");
    } else {
      asm volatile("s_wait_asynccnt 0x0" ::: "memory");
    }
    __syncthreads();

    const unsigned short* sAh = smem;
    const unsigned short* sBh = smem + (curB >> 1);

    // hoist the scalar codebook-norm load: overlaps the whole WMMA chain
    const float en = enorm[ct * CT + r];
    const int   cn = ct * CT + r;

    v8f acc0 = {};
    v8f acc1 = {};
    // two independent row-chains share one B fragment: 6 ds_loads feed 2 WMMAs
    #pragma unroll
    for (int k0 = 0; k0 < DIM; k0 += 32) {
      v8bf blo = *(const v8bf*)&sBh[bColHalf + k0 + kc];
      v8bf bhi = *(const v8bf*)&sBh[bColHalf + k0 + kc + 16];
      v16bf b = __builtin_shufflevector(blo, bhi,
          0, 1, 2, 3, 4, 5, 6, 7, 8, 9, 10, 11, 12, 13, 14, 15);
      v8bf a0lo = *(const v8bf*)&sAh[aRow0Half + k0 + kh];
      v8bf a0hi = *(const v8bf*)&sAh[aRow0Half + k0 + kh + 16];
      v16bf a0 = __builtin_shufflevector(a0lo, a0hi,
          0, 1, 2, 3, 4, 5, 6, 7, 8, 9, 10, 11, 12, 13, 14, 15);
      v8bf a1lo = *(const v8bf*)&sAh[aRow1Half + k0 + kh];
      v8bf a1hi = *(const v8bf*)&sAh[aRow1Half + k0 + kh + 16];
      v16bf a1 = __builtin_shufflevector(a1lo, a1hi,
          0, 1, 2, 3, 4, 5, 6, 7, 8, 9, 10, 11, 12, 13, 14, 15);
      acc0 = __builtin_amdgcn_wmma_f32_16x16x32_bf16(
          false, a0, false, b, (short)0, acc0, false, false);
      acc1 = __builtin_amdgcn_wmma_f32_16x16x32_bf16(
          false, a1, false, b, (short)0, acc1, false, false);
    }

    #pragma unroll
    for (int v = 0; v < 8; ++v) {
      float d0 = en - 2.0f * acc0[v];        // ||e||^2 - 2 z.e (||z||^2 const)
      if (d0 < best0[v]) { best0[v] = d0; bidx0[v] = cn; }
      float d1 = en - 2.0f * acc1[v];
      if (d1 < best1[v]) { best1[v] = d1; bidx1[v] = cn; }
    }
    __syncthreads();                          // tile buffer reused at ct+1
  }

  // reduce across the 16 lanes sharing each row (halves stay separate)
  #pragma unroll
  for (int m = 8; m >= 1; m >>= 1) {
    #pragma unroll
    for (int v = 0; v < 8; ++v) {
      float ob0 = __shfl_xor(best0[v], m, 32);
      int   oi0 = __shfl_xor(bidx0[v], m, 32);
      if (ob0 < best0[v] || (ob0 == best0[v] && oi0 < bidx0[v])) {
        best0[v] = ob0; bidx0[v] = oi0;
      }
      float ob1 = __shfl_xor(best1[v], m, 32);
      int   oi1 = __shfl_xor(bidx1[v], m, 32);
      if (ob1 < best1[v] || (ob1 == best1[v] && oi1 < bidx1[v])) {
        best1[v] = ob1; bidx1[v] = oi1;
      }
    }
  }
  const int rowW = rowBase + wave * 32;
  if (lane == 0) {
    #pragma unroll
    for (int v = 0; v < 8; ++v) {
      idx[rowW + v]      = bidx0[v];
      idx[rowW + 16 + v] = bidx1[v];
    }
  } else if (lane == 16) {
    #pragma unroll
    for (int v = 0; v < 8; ++v) {
      idx[rowW + 8 + v]  = bidx0[v];
      idx[rowW + 24 + v] = bidx1[v];
    }
  }
}

// ------------------------------------------------------------------- epilogue
__global__ __launch_bounds__(256) void vq_gather_kernel(
    const float* __restrict__ z, const float* __restrict__ e,
    const int* __restrict__ idx, float* __restrict__ zq,
    float* __restrict__ counts, float* __restrict__ sse)
{
  const int m = blockIdx.x;
  const int k = idx[m];
  const float4 z4 = ((const float4*)(z + (size_t)m * DIM))[threadIdx.x];
  const float4 e4 = ((const float4*)(e + (size_t)k * DIM))[threadIdx.x];
  float* o = zq + (size_t)m * DIM + (size_t)threadIdx.x * 4;  // 4B-aligned dest
  o[0] = e4.x; o[1] = e4.y; o[2] = e4.z; o[3] = e4.w;
  float dx = z4.x - e4.x, dy = z4.y - e4.y, dz = z4.z - e4.z, dw = z4.w - e4.w;
  float s = dx * dx + dy * dy + dz * dz + dw * dw;
  #pragma unroll
  for (int mm = 16; mm >= 1; mm >>= 1) s += __shfl_xor(s, mm, 32);
  __shared__ float red[8];
  if ((threadIdx.x & 31) == 0) red[threadIdx.x >> 5] = s;
  __syncthreads();
  if (threadIdx.x == 0) {
    float t = 0.f;
    #pragma unroll
    for (int i = 0; i < 8; ++i) t += red[i];
    atomicAdd(sse, t);
    atomicAdd(&counts[k], 1.0f);
  }
}

__global__ __launch_bounds__(256) void vq_finalize_kernel(
    const float* __restrict__ counts, const float* __restrict__ sse,
    float* __restrict__ loss_out, float* __restrict__ ppl_out)
{
  float s = 0.f;
  const float inv = 1.0f / (float)M_TOK;
  for (int i = threadIdx.x; i < N_EMB; i += 256) {
    float em = counts[i] * inv;
    s += em * __logf(em + 1e-10f);
  }
  #pragma unroll
  for (int m = 16; m >= 1; m >>= 1) s += __shfl_xor(s, m, 32);
  __shared__ float red[8];
  if ((threadIdx.x & 31) == 0) red[threadIdx.x >> 5] = s;
  __syncthreads();
  if (threadIdx.x == 0) {
    float t = 0.f;
    #pragma unroll
    for (int i = 0; i < 8; ++i) t += red[i];
    *ppl_out  = __expf(-t);
    *loss_out = (1.0f + 0.25f) * (*sse) / ((float)M_TOK * (float)DIM);
  }
}

// -------------------------------------------------------------------- launch
extern "C" void kernel_launch(void* const* d_in, const int* in_sizes, int n_in,
                              void* d_out, int out_size, void* d_ws, size_t ws_size,
                              hipStream_t stream) {
  (void)in_sizes; (void)n_in; (void)out_size; (void)ws_size;
  const float* z   = (const float*)d_in[0];
  const float* emb = (const float*)d_in[1];
  float* out = (float*)d_out;

  char* w = (char*)d_ws;
  unsigned short* zb = (unsigned short*)w; w += (size_t)M_TOK * DIM * 2;  // 32 MB
  unsigned short* eb = (unsigned short*)w; w += (size_t)N_EMB * DIM * 2;  // 16 MB
  float* enorm  = (float*)w; w += (size_t)N_EMB * 4;
  int*   idxbuf = (int*)w;   w += (size_t)M_TOK * 4;
  float* counts = (float*)w; w += (size_t)N_EMB * 4;
  float* sse    = (float*)w; w += 16;

  // zero counts + sse (adjacent)
  hipMemsetAsync(counts, 0, (size_t)N_EMB * 4 + 16, stream);

  conv_z_kernel<<<4096, 256, 0, stream>>>(z, zb, M_TOK * DIM);
  conv_e_kernel<<<N_EMB, 256, 0, stream>>>(emb, eb, enorm);

  vq_argmin_kernel<<<M_TOK / RPB, BTHREADS, LDS_BYTES, stream>>>(
      zb, eb, enorm, idxbuf);

  float* loss_out = out;
  float* zq_out   = out + 1;
  float* emb_out  = out + 1 + (size_t)M_TOK * DIM;
  float* ppl_out  = out + 1 + (size_t)M_TOK * DIM + (size_t)N_EMB * DIM;

  vq_gather_kernel<<<M_TOK, 256, 0, stream>>>(z, emb, idxbuf, zq_out, counts, sse);
  hipMemcpyAsync(emb_out, emb, (size_t)N_EMB * DIM * sizeof(float),
                 hipMemcpyDeviceToDevice, stream);
  vq_finalize_kernel<<<1, 256, 0, stream>>>(counts, sse, loss_out, ppl_out);
}
